// encoder_85031762526823
// MI455X (gfx1250) — matrix operand order repaired and checked by reading, not compiled, not run on hardware
//
#include <hip/hip_runtime.h>
#include <hip/hip_bf16.h>

// ---------------- problem constants ----------------
#define BATCH  256
#define TSTEPS 127      // T-1
#define DIN    512
#define HID    512
#define NL     3
#define NGATE  2048     // 4*HID
#define KTOT   1024     // DIN + HID (concat [x|h])

typedef __attribute__((ext_vector_type(16))) __bf16 v16bf;
typedef __attribute__((ext_vector_type(8)))  float  v8f;

union FragBF { v16bf v; uint4 u[2]; };

__device__ __forceinline__ float sigmoidf(float x) { return 1.0f / (1.0f + __expf(-x)); }

// ---------------------------------------------------------------------------
// 1) Time-invariant attention: attn[b,d] = softmax_d( sum_t x[b,t,d]*aw[2H+t] )
//    (h/c terms and attn_b are constant across d -> cancel in softmax)
// ---------------------------------------------------------------------------
__global__ void attn_softmax_kernel(const float* __restrict__ x,
                                    const float* __restrict__ aw,
                                    float* __restrict__ attn) {
    __shared__ float red[256];
    const int b   = blockIdx.x;
    const int tid = threadIdx.x;                 // 256 threads, 2 features each
    const float* xb = x + (size_t)b * TSTEPS * DIN;
    float v0 = 0.f, v1 = 0.f;
    for (int t = 0; t < TSTEPS; ++t) {
        const float w = aw[2 * HID + t];
        v0 += xb[(size_t)t * DIN + tid]       * w;
        v1 += xb[(size_t)t * DIN + tid + 256] * w;
    }
    float m = fmaxf(v0, v1);
    red[tid] = m; __syncthreads();
    for (int s = 128; s > 0; s >>= 1) { if (tid < s) red[tid] = fmaxf(red[tid], red[tid + s]); __syncthreads(); }
    m = red[0]; __syncthreads();
    const float e0 = __expf(v0 - m), e1 = __expf(v1 - m);
    red[tid] = e0 + e1; __syncthreads();
    for (int s = 128; s > 0; s >>= 1) { if (tid < s) red[tid] += red[tid + s]; __syncthreads(); }
    const float inv = 1.0f / red[0];
    attn[(size_t)b * DIN + tid]       = e0 * inv;
    attn[(size_t)b * DIN + tid + 256] = e1 * inv;
}

// ---------------------------------------------------------------------------
// 2) weighted[b,t,d] = attn[b,d]*x[b,t,d]  -> fp32 output + bf16 step-major copy
// ---------------------------------------------------------------------------
__global__ void weighted_kernel(const float* __restrict__ x,
                                const float* __restrict__ attn,
                                float* __restrict__ outw,
                                __hip_bfloat16* __restrict__ xbf) {
    const size_t idx = (size_t)blockIdx.x * blockDim.x + threadIdx.x;
    if (idx >= (size_t)BATCH * TSTEPS * DIN) return;
    const int d = (int)(idx % DIN);
    const int t = (int)((idx / DIN) % TSTEPS);
    const int b = (int)(idx / ((size_t)DIN * TSTEPS));
    const float w = attn[(size_t)b * DIN + d] * x[idx];
    outw[idx] = w;
    xbf[((size_t)t * BATCH + b) * DIN + d] = __hip_bfloat16(w);   // [t][b][d]
}

// ---------------------------------------------------------------------------
// 3) Weight prep: Wbf[l][g][k] = bf16( k<512 ? W_ih[l][g][k] : W_hh[l][g][k-512] )
// ---------------------------------------------------------------------------
__global__ void convert_weights(const float* __restrict__ Wih,
                                const float* __restrict__ Whh,
                                __hip_bfloat16* __restrict__ Wbf) {
    const size_t idx = (size_t)blockIdx.x * blockDim.x + threadIdx.x;
    if (idx >= (size_t)NL * NGATE * KTOT) return;
    const int    k  = (int)(idx % KTOT);
    const size_t lg = idx / KTOT;                 // l*NGATE + g
    const float v = (k < DIN) ? Wih[lg * DIN + k] : Whh[lg * HID + (k - DIN)];
    Wbf[idx] = __hip_bfloat16(v);
}

__global__ void bias_sum_kernel(const float* __restrict__ bih,
                                const float* __restrict__ bhh,
                                float* __restrict__ bsum) {
    const int idx = blockIdx.x * blockDim.x + threadIdx.x;
    if (idx < NL * NGATE) bsum[idx] = bih[idx] + bhh[idx];
}

__global__ void init_state_kernel(float* __restrict__ c, __hip_bfloat16* __restrict__ h) {
    const int idx = blockIdx.x * blockDim.x + threadIdx.x;
    if (idx < NL * BATCH * HID) { c[idx] = 0.f; h[idx] = __hip_bfloat16(0.f); }
}

// ---------------------------------------------------------------------------
// 4a) gates[B,4H] = concat(A0[B,512], A1[B,512]) @ W[4H,1024]^T + bias
//     One wave per 16x64 output strip (4 N-tiles): each K-chunk loads ONE
//     A fragment reused across 4 WMMAs (~13 FLOP/B from L2).
//     Explicit double-buffered pipeline: fragments for chunk i+1 are loaded
//     before chunk i's WMMAs issue, so loads fly under the matrix ops.
//     __launch_bounds__(256,1): grid is only 64 blocks, spend VGPRs freely.
// ---------------------------------------------------------------------------
__global__ void __launch_bounds__(256, 1)
gemm_gates_wmma(const __hip_bfloat16* __restrict__ A0,   // layer input  [B][512]
                const __hip_bfloat16* __restrict__ A1,   // h (prev step)[B][512]
                const __hip_bfloat16* __restrict__ W,    // [2048][1024]
                const float* __restrict__ bias,          // [2048]
                float* __restrict__ gates)               // [B][2048]
{
    const int lane   = threadIdx.x & 31;
    const int warp   = threadIdx.x >> 5;              // 0..7
    const int tile_m = blockIdx.x >> 2;               // 0..15  (256/16)
    const int ngrp   = (blockIdx.x & 3) * 8 + warp;   // 0..31
    const int n0     = ngrp * 64;                     // first of 4 N-tiles
    const int hi     = lane >> 4;                     // lane half
    const int l15    = lane & 15;
    const int m      = tile_m * 16 + l15;

    // A layout: lane half 'hi' owns 16 consecutive K at hi*16 (matches B)
    const __hip_bfloat16* arow0 = A0 + (size_t)m * DIN + hi * 16;
    const __hip_bfloat16* arow1 = A1 + (size_t)m * HID + hi * 16;
    // B layout (this code's order, same as A): lane half 'hi' owns 16 consecutive K at hi*16, column = lane&15
    const __hip_bfloat16* wrow  = W + (size_t)(n0 + l15) * KTOT + hi * 16;
    const size_t WN = (size_t)16 * KTOT;              // stride between N-tiles

    auto loadA = [&](int i, FragBF& a) {
        const __hip_bfloat16* ap = (i < 16) ? (arow0 + i * 32) : (arow1 + (i - 16) * 32);
        a.u[0] = *reinterpret_cast<const uint4*>(ap);
        a.u[1] = *reinterpret_cast<const uint4*>(ap + 8);
    };
    auto loadB = [&](int i, FragBF& b0, FragBF& b1, FragBF& b2, FragBF& b3) {
        const __hip_bfloat16* wp = wrow + i * 32;
        b0.u[0] = *reinterpret_cast<const uint4*>(wp);
        b0.u[1] = *reinterpret_cast<const uint4*>(wp + 8);
        b1.u[0] = *reinterpret_cast<const uint4*>(wp + WN);
        b1.u[1] = *reinterpret_cast<const uint4*>(wp + WN + 8);
        b2.u[0] = *reinterpret_cast<const uint4*>(wp + 2 * WN);
        b2.u[1] = *reinterpret_cast<const uint4*>(wp + 2 * WN + 8);
        b3.u[0] = *reinterpret_cast<const uint4*>(wp + 3 * WN);
        b3.u[1] = *reinterpret_cast<const uint4*>(wp + 3 * WN + 8);
    };

    v8f acc0 = {}, acc1 = {}, acc2 = {}, acc3 = {};

    FragBF a, b0, b1, b2, b3;
    loadA(0, a);
    loadB(0, b0, b1, b2, b3);

    #pragma unroll 2
    for (int i = 0; i < 31; ++i) {
        FragBF an, bn0, bn1, bn2, bn3;
        loadA(i + 1, an);                               // next chunk in flight...
        loadB(i + 1, bn0, bn1, bn2, bn3);
        __builtin_prefetch(wrow + (i + 4) * 32, 0, 3);  // WGP-scope W-stream prefetch
        // ...while current chunk computes (4 independent WMMA chains)
        acc0 = __builtin_amdgcn_wmma_f32_16x16x32_bf16(false, a.v, false, b0.v, (short)0, acc0, false, false);
        acc1 = __builtin_amdgcn_wmma_f32_16x16x32_bf16(false, a.v, false, b1.v, (short)0, acc1, false, false);
        acc2 = __builtin_amdgcn_wmma_f32_16x16x32_bf16(false, a.v, false, b2.v, (short)0, acc2, false, false);
        acc3 = __builtin_amdgcn_wmma_f32_16x16x32_bf16(false, a.v, false, b3.v, (short)0, acc3, false, false);
        a = an; b0 = bn0; b1 = bn1; b2 = bn2; b3 = bn3;
    }
    acc0 = __builtin_amdgcn_wmma_f32_16x16x32_bf16(false, a.v, false, b0.v, (short)0, acc0, false, false);
    acc1 = __builtin_amdgcn_wmma_f32_16x16x32_bf16(false, a.v, false, b1.v, (short)0, acc1, false, false);
    acc2 = __builtin_amdgcn_wmma_f32_16x16x32_bf16(false, a.v, false, b2.v, (short)0, acc2, false, false);
    acc3 = __builtin_amdgcn_wmma_f32_16x16x32_bf16(false, a.v, false, b3.v, (short)0, acc3, false, false);

    // C/D layout: N = lane&15, M = r + 8*hi
    const int mrow = tile_m * 16 + hi * 8;
    float* gbase = gates + (size_t)mrow * NGATE + n0 + l15;
    {
        const float bv = bias[n0 + l15];
        #pragma unroll
        for (int r = 0; r < 8; ++r) gbase[(size_t)r * NGATE] = acc0[r] + bv;
    }
    {
        const float bv = bias[n0 + 16 + l15];
        #pragma unroll
        for (int r = 0; r < 8; ++r) gbase[(size_t)r * NGATE + 16] = acc1[r] + bv;
    }
    {
        const float bv = bias[n0 + 32 + l15];
        #pragma unroll
        for (int r = 0; r < 8; ++r) gbase[(size_t)r * NGATE + 32] = acc2[r] + bv;
    }
    {
        const float bv = bias[n0 + 48 + l15];
        #pragma unroll
        for (int r = 0; r < 8; ++r) gbase[(size_t)r * NGATE + 48] = acc3[r] + bv;
    }
}

// ---------------------------------------------------------------------------
// 4b) LSTM elementwise: gate order (i,f,g,o); updates c (fp32), h (bf16);
//     layer 2 also writes input_encoded[b][t][j] (enc pre-offset by t*HID).
// ---------------------------------------------------------------------------
__global__ void lstm_pointwise(const float* __restrict__ gates,
                               float* __restrict__ c,
                               __hip_bfloat16* __restrict__ h,
                               float* __restrict__ enc) {
    const int idx = blockIdx.x * blockDim.x + threadIdx.x;    // BATCH*HID
    if (idx >= BATCH * HID) return;
    const int b = idx >> 9;
    const int j = idx & (HID - 1);
    const float* g = gates + (size_t)b * NGATE;
    const float ig = sigmoidf(g[j]);
    const float fg = sigmoidf(g[HID + j]);
    const float gg = tanhf  (g[2 * HID + j]);
    const float og = sigmoidf(g[3 * HID + j]);
    const float cn = fg * c[idx] + ig * gg;
    const float hn = og * tanhf(cn);
    c[idx] = cn;
    h[idx] = __hip_bfloat16(hn);
    if (enc) enc[(size_t)b * TSTEPS * HID + j] = hn;
}

// ---------------------------------------------------------------------------
extern "C" void kernel_launch(void* const* d_in, const int* in_sizes, int n_in,
                              void* d_out, int out_size, void* d_ws, size_t ws_size,
                              hipStream_t stream) {
    const float* x   = (const float*)d_in[0];   // [256][127][512]
    const float* Wih = (const float*)d_in[1];   // [3][2048][512]
    const float* Whh = (const float*)d_in[2];   // [3][2048][512]
    const float* bih = (const float*)d_in[3];   // [3][2048]
    const float* bhh = (const float*)d_in[4];   // [3][2048]
    const float* aw  = (const float*)d_in[5];   // [1151][1]
    // d_in[6] = attn_b: constant across softmax axis -> cancels, unused.

    float* out_w   = (float*)d_out;                                   // [B][T-1][D]
    float* out_enc = out_w + (size_t)BATCH * TSTEPS * DIN;            // [B][T-1][H]

    // ---- workspace carve-out (256B aligned) ----
    char*  ws  = (char*)d_ws;
    size_t off = 0;
    auto take = [&](size_t bytes) -> char* {
        char* p = ws + off;
        off = (off + bytes + 255) & ~(size_t)255;
        return p;
    };
    float*          attn  = (float*)         take((size_t)BATCH * DIN * 4);
    __hip_bfloat16* xbf   = (__hip_bfloat16*)take((size_t)TSTEPS * BATCH * DIN * 2);
    __hip_bfloat16* Wbf   = (__hip_bfloat16*)take((size_t)NL * NGATE * KTOT * 2);
    float*          bsum  = (float*)         take((size_t)NL * NGATE * 4);
    __hip_bfloat16* hbuf  = (__hip_bfloat16*)take((size_t)NL * BATCH * HID * 2);
    float*          cbuf  = (float*)         take((size_t)NL * BATCH * HID * 4);
    float*          gates = (float*)         take((size_t)BATCH * NGATE * 4);
    (void)ws_size; (void)in_sizes; (void)n_in; (void)out_size;

    // ---- one-time prep ----
    attn_softmax_kernel<<<BATCH, 256, 0, stream>>>(x, aw, attn);

    {
        const size_t total = (size_t)BATCH * TSTEPS * DIN;
        weighted_kernel<<<(unsigned)((total + 255) / 256), 256, 0, stream>>>(x, attn, out_w, xbf);
    }
    {
        const size_t total = (size_t)NL * NGATE * KTOT;
        convert_weights<<<(unsigned)((total + 255) / 256), 256, 0, stream>>>(Wih, Whh, Wbf);
    }
    bias_sum_kernel<<<(NL * NGATE + 255) / 256, 256, 0, stream>>>(bih, bhh, bsum);
    init_state_kernel<<<(NL * BATCH * HID + 255) / 256, 256, 0, stream>>>(cbuf, hbuf);

    // ---- recurrent loop: 127 steps x 3 layers ----
    const size_t BH = (size_t)BATCH * HID;
    for (int t = 0; t < TSTEPS; ++t) {
        for (int l = 0; l < NL; ++l) {
            const __hip_bfloat16* A0 =
                (l == 0) ? (xbf + (size_t)t * BATCH * DIN)   // attn-weighted input
                         : (hbuf + (size_t)(l - 1) * BH);    // lower layer's fresh h
            gemm_gates_wmma<<<64, 256, 0, stream>>>(
                A0, hbuf + (size_t)l * BH,
                Wbf + (size_t)l * NGATE * KTOT,
                bsum + (size_t)l * NGATE, gates);
            lstm_pointwise<<<(BATCH * HID + 255) / 256, 256, 0, stream>>>(
                gates, cbuf + (size_t)l * BH, hbuf + (size_t)l * BH,
                (l == NL - 1) ? (out_enc + (size_t)t * HID) : nullptr);
        }
    }
}
